// NodeProcessor_6133213298854
// MI455X (gfx1250) — matrix operand order repaired and checked by reading, not compiled, not run on hardware
//
#include <hip/hip_runtime.h>

#define DDIM 128
#define HDIM 256
#define LN_EPS 1e-5f

typedef __attribute__((ext_vector_type(16))) __bf16 v16bf;
typedef __attribute__((ext_vector_type(8)))  float  v8f;

union BFrag { v16bf v; unsigned int u[8]; };

__device__ __forceinline__ unsigned short f2bf(float a) {
    unsigned int ua = __float_as_uint(a);
    ua += 0x7FFFu + ((ua >> 16) & 1u);          // round-to-nearest-even
    return (unsigned short)(ua >> 16);
}

__device__ __forceinline__ unsigned int f2bf_pack(float a, float b) {
    unsigned int ua = __float_as_uint(a);
    ua += 0x7FFFu + ((ua >> 16) & 1u);
    unsigned int ub = __float_as_uint(b);
    ub += 0x7FFFu + ((ub >> 16) & 1u);
    return (ua >> 16) | (ub & 0xFFFF0000u);
}

// ---------------------------------------------------------------------------
// Convert + transpose weights to bf16 so WMMA B-fragments are contiguous:
// W1t[n][k] (256x256), W2t[n][k] (128x256)
// ---------------------------------------------------------------------------
__global__ void cvt_weights_kernel(const float* __restrict__ W1,
                                   const float* __restrict__ W2,
                                   unsigned short* __restrict__ W1t,
                                   unsigned short* __restrict__ W2t) {
    int t = blockIdx.x * 256 + threadIdx.x;
    if (t < 256 * 256) {
        int n = t & 255, k = t >> 8;            // W1 is [k=256][n=256]
        W1t[n * 256 + k] = f2bf(W1[k * 256 + n]);
    } else {
        int t2 = t - 256 * 256;
        if (t2 < 256 * 128) {
            int n = t2 & 127, k = t2 >> 7;      // W2 is [k=256][n=128]
            W2t[n * 256 + k] = f2bf(W2[k * 128 + n]);
        }
    }
}

// ---------------------------------------------------------------------------
// Edge scatter: agg[dst] += edge_attr[e]; 32 lanes per edge, float4 per lane
// ---------------------------------------------------------------------------
__global__ void scatter_edges_kernel(const float* __restrict__ edge_attr,
                                     const int* __restrict__ edge_index,
                                     float* __restrict__ agg, int nEdges) {
    long long gid = (long long)blockIdx.x * blockDim.x + threadIdx.x;
    int e = (int)(gid >> 5);
    if (e >= nEdges) return;
    int j = (int)(gid & 31);
    float4 v = *(const float4*)(edge_attr + (long long)e * DDIM + j * 4);
    int dst = edge_index[e];                     // row 0 of [2, E]
    float* p = agg + (long long)dst * DDIM + j * 4;
    unsafeAtomicAdd(p + 0, v.x);
    unsafeAtomicAdd(p + 1, v.y);
    unsafeAtomicAdd(p + 2, v.z);
    unsafeAtomicAdd(p + 3, v.w);
}

// ---------------------------------------------------------------------------
// Fused MLP: [x|agg] -> (W1,b1,SiLU) -> (W2,b2) -> LayerNorm -> +x
// Block = 128 nodes, 256 threads (8 waves), wave w owns rows w*16..w*16+15.
// ---------------------------------------------------------------------------
__global__ __launch_bounds__(256) void fused_mlp_kernel(
    const float* __restrict__ x, const float* __restrict__ agg,
    const unsigned short* __restrict__ W1t, const float* __restrict__ b1,
    const unsigned short* __restrict__ W2t, const float* __restrict__ b2,
    const float* __restrict__ gamma, const float* __restrict__ beta,
    float* __restrict__ out, int nNodes) {

    __shared__ unsigned short hMid[128 * 256];   // 64 KB, bf16, XOR-swizzled

    const int tid   = threadIdx.x;
    const int wv    = tid >> 5;
    const int lane  = tid & 31;
    const int lhalf = lane >> 4;                 // K-half for A, M-half for C/D
    const int l16   = lane & 15;
    const int rowLocal = wv * 16 + l16;          // A-matrix row for this lane
    const int rowG  = blockIdx.x * 128 + rowLocal;
    const int rowC  = min(rowG, nNodes - 1);

    // ---- A fragments for GEMM1 from global x | agg (fp32 -> bf16) ----
    // ISA A layout (16-bit, 16x32): u[p] p<4 -> k = half*8+2p ; p>=4 -> k = 16+half*8+2(p-4)
    BFrag afrag[8];
    const float* xr = x   + (long long)rowC * DDIM;
    const float* ar = agg + (long long)rowC * DDIM;
#pragma unroll
    for (int kb = 0; kb < 8; ++kb) {
        const float* base = (kb < 4) ? (xr + kb * 32) : (ar + (kb - 4) * 32);
#pragma unroll
        for (int p = 0; p < 4; ++p) {
            int k0 = lhalf * 8 + 2 * p;
            float2 lo = *(const float2*)(base + k0);
            float2 hi = *(const float2*)(base + 16 + k0);
            afrag[kb].u[p]     = f2bf_pack(lo.x, lo.y);
            afrag[kb].u[4 + p] = f2bf_pack(hi.x, hi.y);
        }
    }

    // ---- GEMM1 (128x256 @ 256x256) + bias + SiLU -> LDS bf16 ----
    for (int nt = 0; nt < 16; ++nt) {
        int n = nt * 16 + l16;
        float bias = b1[n];
        v8f acc = {bias, bias, bias, bias, bias, bias, bias, bias};
#pragma unroll
        for (int kb = 0; kb < 8; ++kb) {
            // B layout: lane n holds k = kb*32 + lanehalf*8 + j (j<8), kb*32 + 16 + lanehalf*8 + (j-8) (j>=8)
            const uint4* wp = (const uint4*)(W1t + n * 256 + kb * 32 + lhalf * 8);
            uint4 q0 = wp[0], q1 = wp[2];
            BFrag bf;
            bf.u[0] = q0.x; bf.u[1] = q0.y; bf.u[2] = q0.z; bf.u[3] = q0.w;
            bf.u[4] = q1.x; bf.u[5] = q1.y; bf.u[6] = q1.z; bf.u[7] = q1.w;
            acc = __builtin_amdgcn_wmma_f32_16x16x32_bf16(
                false, afrag[kb].v, false, bf.v, (short)0, acc, false, false);
        }
#pragma unroll
        for (int r = 0; r < 8; ++r) {            // C/D: VGPR r -> row r + 8*half
            float v = acc[r];
            float s = v / (1.0f + __expf(-v));   // SiLU
            int m = wv * 16 + r + 8 * lhalf;
            int w = (n >> 1) ^ ((m & 15) * 4);   // bank swizzle in u32-word space
            hMid[m * 256 + w * 2 + (n & 1)] = f2bf(s);
        }
    }
    __syncthreads();

    // ---- GEMM2 A fragments from LDS (wave reads only rows it wrote) ----
    BFrag a2[8];
#pragma unroll
    for (int kb = 0; kb < 8; ++kb) {
#pragma unroll
        for (int p = 0; p < 8; ++p) {
            int k0 = (p < 4) ? (lhalf * 8 + 2 * p) : (16 + lhalf * 8 + 2 * (p - 4));
            int w  = ((kb * 32 + k0) >> 1) ^ ((rowLocal & 15) * 4);
            a2[kb].u[p] = *(const unsigned int*)&hMid[rowLocal * 256 + w * 2];
        }
    }

    // ---- GEMM2 (128x256 @ 256x128) + bias ----
    v8f acc2[8];
#pragma unroll
    for (int nt = 0; nt < 8; ++nt) {
        int n = nt * 16 + l16;
        float bias = b2[n];
        v8f acc = {bias, bias, bias, bias, bias, bias, bias, bias};
#pragma unroll
        for (int kb = 0; kb < 8; ++kb) {
            const uint4* wp = (const uint4*)(W2t + n * 256 + kb * 32 + lhalf * 8);
            uint4 q0 = wp[0], q1 = wp[2];
            BFrag bf;
            bf.u[0] = q0.x; bf.u[1] = q0.y; bf.u[2] = q0.z; bf.u[3] = q0.w;
            bf.u[4] = q1.x; bf.u[5] = q1.y; bf.u[6] = q1.z; bf.u[7] = q1.w;
            acc = __builtin_amdgcn_wmma_f32_16x16x32_bf16(
                false, a2[kb].v, false, bf.v, (short)0, acc, false, false);
        }
        acc2[nt] = acc;
    }

    // ---- LayerNorm (over 128 cols of each row) + residual + store ----
    float garr[8], barr[8];
#pragma unroll
    for (int nt = 0; nt < 8; ++nt) {
        int n = nt * 16 + l16;
        garr[nt] = gamma[n];
        barr[nt] = beta[n];
    }
#pragma unroll
    for (int r = 0; r < 8; ++r) {
        float s = 0.f, ss = 0.f;
#pragma unroll
        for (int nt = 0; nt < 8; ++nt) { float v = acc2[nt][r]; s += v; ss += v * v; }
        // row lives on 16 lanes of one half -> width-16 butterfly reduce
#pragma unroll
        for (int msk = 1; msk < 16; msk <<= 1) {
            s  += __shfl_xor(s,  msk, 16);
            ss += __shfl_xor(ss, msk, 16);
        }
        float mu  = s * (1.0f / 128.0f);
        float var = ss * (1.0f / 128.0f) - mu * mu;
        float inv = rsqrtf(var + LN_EPS);

        int  mG = blockIdx.x * 128 + wv * 16 + r + 8 * lhalf;
        bool ok = mG < nNodes;
        int  mC = ok ? mG : (nNodes - 1);
#pragma unroll
        for (int nt = 0; nt < 8; ++nt) {
            int n = nt * 16 + l16;
            float xv = x[(long long)mC * DDIM + n];
            float v  = (acc2[nt][r] - mu) * inv * garr[nt] + barr[nt] + xv;
            if (ok) out[(long long)mG * DDIM + n] = v;
        }
    }
}

// ---------------------------------------------------------------------------
extern "C" void kernel_launch(void* const* d_in, const int* in_sizes, int n_in,
                              void* d_out, int out_size, void* d_ws, size_t ws_size,
                              hipStream_t stream) {
    const float* x     = (const float*)d_in[0];
    const int*   eidx  = (const int*)  d_in[1];   // edge_index [2, E], row 0 = dst
    const float* eattr = (const float*)d_in[2];
    const float* W1    = (const float*)d_in[3];
    const float* b1    = (const float*)d_in[4];
    const float* W2    = (const float*)d_in[5];
    const float* b2    = (const float*)d_in[6];
    const float* gam   = (const float*)d_in[7];
    const float* bet   = (const float*)d_in[8];

    int nNodes = in_sizes[0] / DDIM;
    int nEdges = in_sizes[2] / DDIM;

    // workspace layout: [agg fp32 N*D][W1t bf16 256*256][W2t bf16 128*256]
    float* agg = (float*)d_ws;
    size_t aggBytes = (size_t)nNodes * DDIM * sizeof(float);
    size_t off = (aggBytes + 255) & ~(size_t)255;
    unsigned short* W1t = (unsigned short*)((char*)d_ws + off);
    unsigned short* W2t = (unsigned short*)((char*)d_ws + off + 256 * 256 * 2);

    hipMemsetAsync(agg, 0, aggBytes, stream);

    cvt_weights_kernel<<<(256 * 256 + 256 * 128) / 256, 256, 0, stream>>>(W1, W2, W1t, W2t);

    long long sthreads = (long long)nEdges * 32;
    scatter_edges_kernel<<<(int)((sthreads + 255) / 256), 256, 0, stream>>>(eattr, eidx, agg, nEdges);

    int blocks = (nNodes + 127) / 128;
    fused_mlp_kernel<<<blocks, 256, 0, stream>>>(x, agg, W1t, b1, W2t, b2, gam, bet,
                                                 (float*)d_out, nNodes);
}
